// DGNet_73375221285247
// MI455X (gfx1250) — matrix operand order repaired and checked by reading, not compiled, not run on hardware
//
#include <hip/hip_runtime.h>
#include <hip/hip_bf16.h>

// ---------------------------------------------------------------------------
// DGNet (bilateral-grid net) for MI455X / gfx1250.
// All intermediate activations are NHWC bf16 (numerically identical to
// converting at the consumer, halves HBM traffic, and makes the LDS halo a
// contiguous byte range -> staged with global_load_async_to_lds_b128).
//  conv0 (3->8)                : scalar, packed b128 NHWC-bf16 stores
//  conv1..4, feat0..3, outconv : implicit-GEMM, v_wmma_f32_16x16x32_bf16
//  bilateral slice + apply     : scalar bandwidth kernel (f32 grid, float2)
// ---------------------------------------------------------------------------

typedef __bf16 bf16_t;
typedef bf16_t v16bf __attribute__((ext_vector_type(16)));
typedef float  v8f   __attribute__((ext_vector_type(8)));

union Frag {
    v16bf        v;
    unsigned int d[8];
};

__device__ __forceinline__ unsigned short f2bf(float f) {
    unsigned int u = __float_as_uint(f);
    u += 0x7FFFu + ((u >> 16) & 1u);      // round to nearest even
    return (unsigned short)(u >> 16);
}

// CDNA5 async global->LDS copies, tracked by ASYNCcnt.
__device__ __forceinline__ void async_copy_b32(const float* gp, float* lp) {
    unsigned int lds_off = (unsigned int)(unsigned long long)lp; // addr[31:0]
    asm volatile("global_load_async_to_lds_b32 %0, %1, off"
                 :: "v"(lds_off), "v"((unsigned long long)gp)
                 : "memory");
}
__device__ __forceinline__ void async_copy_b128(const unsigned short* gp,
                                                unsigned short* lp) {
    unsigned int lds_off = (unsigned int)(unsigned long long)lp;
    asm volatile("global_load_async_to_lds_b128 %0, %1, off"
                 :: "v"(lds_off), "v"((unsigned long long)gp)
                 : "memory");
}
__device__ __forceinline__ void wait_asynccnt0() {
    asm volatile("s_wait_asynccnt 0x0" ::: "memory");
}

// ---------------------------------------------------------------------------
// conv0: 3 planes (ref, masked, mask, NCHW f32) -> 8 ch, 3x3, s2, pad 1, relu.
// Output NHWC bf16: one 16-byte store per pixel.
// ---------------------------------------------------------------------------
template<int OH>
__global__ __launch_bounds__(256)
void conv0_kernel(const float* __restrict__ ref,
                  const float* __restrict__ msk,
                  const float* __restrict__ mmm,
                  const float* __restrict__ w,         // (8,3,3,3)
                  const float* __restrict__ bias,      // (8)
                  unsigned short* __restrict__ out)    // NHWC bf16 (B,OH,OH,8)
{
    __shared__ float sw[8 * 3 * 9];
    __shared__ float sb[8];
    const int tid = threadIdx.x;
    if (tid < 8 * 3 * 9) sw[tid] = w[tid];
    if (tid < 8)         sb[tid] = bias[tid];
    __syncthreads();

    constexpr int IH = OH * 2;
    const int idx = blockIdx.x * 256 + tid;          // over B*OH*OH (exact)
    const int b   = idx / (OH * OH);                 // pow2 -> shifts
    const int rem = idx - b * (OH * OH);
    const int oy  = rem / OH;
    const int ox  = rem & (OH - 1);

    float acc[8];
#pragma unroll
    for (int c = 0; c < 8; ++c) acc[c] = sb[c];

#pragma unroll
    for (int ky = 0; ky < 3; ++ky) {
        const int iy = oy * 2 + ky - 1;
        if (iy < 0 || iy >= IH) continue;
#pragma unroll
        for (int kx = 0; kx < 3; ++kx) {
            const int ix = ox * 2 + kx - 1;
            if (ix < 0 || ix >= IH) continue;
            const int off = (b * IH + iy) * IH + ix;
            const float v0 = ref[off];
            const float v1 = msk[off];
            const float v2 = mmm[off];
            const int tap  = ky * 3 + kx;
#pragma unroll
            for (int c = 0; c < 8; ++c) {
                acc[c] = fmaf(sw[(c * 3 + 0) * 9 + tap], v0,
                         fmaf(sw[(c * 3 + 1) * 9 + tap], v1,
                         fmaf(sw[(c * 3 + 2) * 9 + tap], v2, acc[c])));
            }
        }
    }
    uint4 pk;
    pk.x = (unsigned)f2bf(fmaxf(acc[0], 0.f)) | ((unsigned)f2bf(fmaxf(acc[1], 0.f)) << 16);
    pk.y = (unsigned)f2bf(fmaxf(acc[2], 0.f)) | ((unsigned)f2bf(fmaxf(acc[3], 0.f)) << 16);
    pk.z = (unsigned)f2bf(fmaxf(acc[4], 0.f)) | ((unsigned)f2bf(fmaxf(acc[5], 0.f)) << 16);
    pk.w = (unsigned)f2bf(fmaxf(acc[6], 0.f)) | ((unsigned)f2bf(fmaxf(acc[7], 0.f)) << 16);
    *(uint4*)&out[(long)idx * 8] = pk;               // NHWC, 16B aligned
}

// ---------------------------------------------------------------------------
// Implicit-GEMM conv via WMMA bf16. Input NHWC bf16, output NHWC (bf16 or f32).
//   A (16x32) = weights    : M = 16 output channels, K = Cin*KS*KS
//   B (32x16) = activation : K,                      N = 16 output pixels
//   D (16x16) = output     : lane = pixel; 8 consecutive channels per lane
//                            -> one b128 store per (nt, half)
// Input halo [ry][rx][ci] staged once per block: interior rows are contiguous
// NHWC bytes -> global_load_async_to_lds_b128 DMA; OOB rows zero-filled;
// edge blocks use a per-element path (all branches block/row-uniform).
// ---------------------------------------------------------------------------
template<int CIN, int COUT, int STRIDE, int KS, int OH, int WAVES,
         bool RELU, bool HASBIAS, bool WASYNC, bool OUTBF16>
__global__ __launch_bounds__(WAVES * 32)
void conv_wmma_kernel(const unsigned short* __restrict__ in,  // NHWC bf16
                      const float* __restrict__ w,            // (COUT,CIN,KS,KS)
                      const float* __restrict__ bias,         // (COUT) or null
                      void* __restrict__ outv)                // NHWC
{
    constexpr int KTOT  = CIN * KS * KS;
    constexpr int KT    = (KTOT + 31) / 32;
    constexpr int NT    = COUT / 16;
    constexpr int IH    = OH * STRIDE;
    constexpr int PAD   = (KS == 3) ? 1 : 0;
    constexpr int BPIX  = WAVES * 16;                  // <= OH, divides OH
    constexpr int SROWS = KS;
    constexpr int SCOLS = (BPIX - 1) * STRIDE + KS;
    constexpr int NTHR  = WAVES * 32;
    constexpr int ROWE  = SCOLS * CIN;                 // halves per staged row

    __shared__ __align__(16) unsigned short s_in[SROWS * ROWE];  // [ry][rx][ci]
    __shared__ unsigned short s_wb[WASYNC ? 2 : COUT * 32];      // bf16 weights
    __shared__ __align__(16) float s_wf[WASYNC ? COUT * 32 : 2]; // f32 weights

    const int tid  = threadIdx.x;
    const int lane = tid & 31;
    const int wave = tid >> 5;
    const int half = (lane >> 4) & 1;
    const int nl   = lane & 15;

    constexpr int HW = OH * OH;
    const int P0    = blockIdx.x * BPIX;
    const int b     = P0 / HW;                         // pow2 -> shift
    const int rem   = P0 - b * HW;
    const int oy    = rem / OH;                        // shared by all waves
    const int oxBlk = rem & (OH - 1);
    const int ox0   = oxBlk + wave * 16;

    const int iy0 = oy * STRIDE - PAD;
    const int ix0 = oxBlk * STRIDE - PAD;

    // --- stage activation halo once (covers ALL K-tiles) ------------------
    const unsigned short* __restrict__ inb = in + (long)b * IH * IH * CIN;
    const bool interiorX = (ix0 >= 0) && (ix0 + SCOLS <= IH);
#pragma unroll
    for (int ry = 0; ry < SROWS; ++ry) {
        const int iy = iy0 + ry;
        unsigned short* dst = &s_in[ry * ROWE];
        if (iy < 0 || iy >= IH) {
            for (int t = tid; t < ROWE / 2; t += NTHR)
                ((unsigned int*)dst)[t] = 0;                    // zero row
        } else if (interiorX) {
            const unsigned short* src = inb + ((long)iy * IH + ix0) * CIN;
            for (int c = tid; c < ROWE / 8; c += NTHR)          // 16B chunks
                async_copy_b128(src + c * 8, dst + c * 8);      // ASYNCcnt DMA
        } else {
            for (int t = tid; t < ROWE; t += NTHR) {
                const int rx = t / CIN;                          // pow2
                const int ci = t & (CIN - 1);
                const int ix = ix0 + rx;
                dst[t] = (ix >= 0 && ix < IH)
                           ? inb[((long)iy * IH + ix) * CIN + ci] : 0;
            }
        }
    }
    wait_asynccnt0();                                  // own wave's DMAs done

    v8f acc[NT];
#pragma unroll
    for (int nt = 0; nt < NT; ++nt)
#pragma unroll
        for (int r = 0; r < 8; ++r) acc[nt][r] = 0.f;

    for (int kt = 0; kt < KT; ++kt) {
        __syncthreads();                               // also covers staging
        // --- stage weight K-tile (32 x COUT) ------------------------------
        for (int idx = tid; idx < COUT * 32; idx += NTHR) {
            const int co = idx >> 5;
            const int kl = idx & 31;
            const int kk = kt * 32 + kl;
            if (WASYNC) {
                if (kk < KTOT) {
                    const int tap = kk / CIN;          // CIN pow2 -> shift
                    const int ci  = kk & (CIN - 1);
                    async_copy_b32(w + (co * CIN + ci) * (KS * KS) + tap,
                                   &s_wf[idx]);
                } else {
                    s_wf[idx] = 0.f;
                }
            } else {
                unsigned short hv = 0;
                if (kk < KTOT) {
                    const int tap = kk / CIN;
                    const int ci  = kk & (CIN - 1);
                    hv = f2bf(w[(co * CIN + ci) * (KS * KS) + tap]);
                }
                s_wb[idx] = hv;
            }
        }
        if (WASYNC) wait_asynccnt0();
        __syncthreads();

        // --- B fragment: one aligned 32-bit LDS read per K-pair -----------
        Frag bF;
#pragma unroll
        for (int v = 0; v < 8; ++v) {
            const int kk = kt * 32 + half * 16 + 2 * v;
            unsigned int dw = 0;
            if (kk < KTOT) {                           // pair-safe (KTOT even)
                const int tap = kk / CIN;
                const int ci  = kk & (CIN - 1);
                int ry, rx;
                if (KS == 3) {
                    const int ky = tap / 3;
                    const int kx = tap - ky * 3;
                    ry = ky;
                    rx = (wave * 16 + nl) * STRIDE + kx;
                } else {
                    ry = 0;
                    rx = wave * 16 + nl;
                }
                dw = *(const unsigned int*)&s_in[(ry * SCOLS + rx) * CIN + ci];
            }
            bF.d[v] = dw;
        }

        // --- A fragments (weights) + WMMA ---------------------------------
#pragma unroll
        for (int nt = 0; nt < NT; ++nt) {
            const int co = nt * 16 + nl;
            Frag aF;
#pragma unroll
            for (int j = 0; j < 8; ++j) {
                const int k0 = half * 16 + 2 * j;
                if (WASYNC) {
                    const float2 f2 = *(const float2*)&s_wf[co * 32 + k0];
                    aF.d[j] = (unsigned int)f2bf(f2.x) |
                              ((unsigned int)f2bf(f2.y) << 16);
                } else {
                    aF.d[j] = *(const unsigned int*)&s_wb[co * 32 + k0];
                }
            }
            acc[nt] = __builtin_amdgcn_wmma_f32_16x16x32_bf16(
                false, aF.v, false, bF.v, (short)0, acc[nt], false, false);
        }
    }

    // --- store NHWC: lane = pixel; 8 consecutive channels per (nt, half) --
    const int pix = (b * OH + oy) * OH + ox0 + nl;
#pragma unroll
    for (int nt = 0; nt < NT; ++nt) {
        float vv[8];
#pragma unroll
        for (int r = 0; r < 8; ++r) {
            const int co = nt * 16 + half * 8 + r;
            float val = acc[nt][r];
            if (HASBIAS) val += bias[co];
            if (RELU)    val = fmaxf(val, 0.f);
            vv[r] = val;
        }
        const int coff = (long)pix * COUT + nt * 16 + half * 8;
        if (OUTBF16) {
            uint4 pk;
            pk.x = (unsigned)f2bf(vv[0]) | ((unsigned)f2bf(vv[1]) << 16);
            pk.y = (unsigned)f2bf(vv[2]) | ((unsigned)f2bf(vv[3]) << 16);
            pk.z = (unsigned)f2bf(vv[4]) | ((unsigned)f2bf(vv[5]) << 16);
            pk.w = (unsigned)f2bf(vv[6]) | ((unsigned)f2bf(vv[7]) << 16);
            *(uint4*)&((unsigned short*)outv)[coff] = pk;        // 16B store
        } else {
            float* op = (float*)outv + coff;
            *(float4*)op       = make_float4(vv[0], vv[1], vv[2], vv[3]);
            *(float4*)(op + 4) = make_float4(vv[4], vv[5], vv[6], vv[7]);
        }
    }
}

// ---------------------------------------------------------------------------
// Bilateral slice + apply at 2048x2048. grid: NHWC f32 (B,64,64,16),
// channel = zi*2 + n -> the two coefficients are an adjacent float2.
// ---------------------------------------------------------------------------
__global__ __launch_bounds__(256)
void slice_apply_kernel(const float* __restrict__ ref,
                        const float* __restrict__ grid,
                        float* __restrict__ out)
{
    const int idx = blockIdx.x * 256 + threadIdx.x;   // over B*2048*2048
    const int y   = (idx >> 11) & 2047;
    const int x   = idx & 2047;
    const int b   = idx >> 22;

    const float g  = ref[idx];
    const float gy = (y + 0.5f) * (1.0f / 32.0f) - 0.5f;
    const float gx = (x + 0.5f) * (1.0f / 32.0f) - 0.5f;
    const float gz = g * 8.0f - 0.5f;

    const float y0f = floorf(gy); const float fy = gy - y0f; const int y0 = (int)y0f;
    const float x0f = floorf(gx); const float fx = gx - x0f; const int x0 = (int)x0f;
    const float z0f = floorf(gz); const float fz = gz - z0f; const int z0 = (int)z0f;

    float c0 = 0.f, c1 = 0.f;
#pragma unroll
    for (int dz = 0; dz < 2; ++dz) {
        const int   zi = min(max(z0 + dz, 0), 7);
        const float wz = dz ? fz : 1.f - fz;
#pragma unroll
        for (int dy = 0; dy < 2; ++dy) {
            const int   yi = min(max(y0 + dy, 0), 63);
            const float wy = dy ? fy : 1.f - fy;
#pragma unroll
            for (int dx = 0; dx < 2; ++dx) {
                const int   xi  = min(max(x0 + dx, 0), 63);
                const float wx  = dx ? fx : 1.f - fx;
                const float wgt = wz * wy * wx;
                const float2 gv =
                    *(const float2*)&grid[((b * 4096 + yi * 64 + xi) << 4) + zi * 2];
                c0 = fmaf(wgt, gv.x, c0);
                c1 = fmaf(wgt, gv.y, c1);
            }
        }
    }
    out[idx] = fmaf(g, c0, c1);
}

// ---------------------------------------------------------------------------
extern "C" void kernel_launch(void* const* d_in, const int* in_sizes, int n_in,
                              void* d_out, int out_size, void* d_ws, size_t ws_size,
                              hipStream_t stream)
{
    const float* ref  = (const float*)d_in[0];
    const float* mski = (const float*)d_in[1];
    const float* mask = (const float*)d_in[2];
    const float* dw0 = (const float*)d_in[3];  const float* db0 = (const float*)d_in[4];
    const float* dw1 = (const float*)d_in[5];  const float* db1 = (const float*)d_in[6];
    const float* dw2 = (const float*)d_in[7];  const float* db2 = (const float*)d_in[8];
    const float* dw3 = (const float*)d_in[9];  const float* db3 = (const float*)d_in[10];
    const float* dw4 = (const float*)d_in[11]; const float* db4 = (const float*)d_in[12];
    const float* fw0 = (const float*)d_in[13];
    const float* fw1 = (const float*)d_in[14];
    const float* fw2 = (const float*)d_in[15];
    const float* fw3 = (const float*)d_in[16];
    const float* fb0 = (const float*)d_in[17];
    const float* fb1 = (const float*)d_in[18];
    const float* fb2 = (const float*)d_in[19];
    const float* ow  = (const float*)d_in[20];
    const float* ob  = (const float*)d_in[21];
    float* outp = (float*)d_out;

    // Workspace (bf16 ping-pong): P = 2*1024*1024*8 halves (33.5MB, conv0 out),
    // Q = 2*512*512*16 halves (16.8MB), R = f32 grid (512KB). Total ~51MB.
    unsigned short* P = (unsigned short*)d_ws;
    unsigned short* Q = P + 2 * 1024 * 1024 * 8;
    float*          R = (float*)(Q + 2 * 512 * 512 * 16);

    // conv0: 3 -> 8, 2048 -> 1024, NHWC bf16 out
    conv0_kernel<1024><<<dim3(2 * 1024 * 1024 / 256), dim3(256), 0, stream>>>(
        ref, mski, mask, dw0, db0, P);

    // stride-2 WMMA convs; blocks = B*OH*OH / (WAVES*16)
    conv_wmma_kernel<  8,  16, 2, 3, 512, 8, true, true, true , true>
        <<<dim3(4096), dim3(256), 0, stream>>>(P, dw1, db1, Q);
    conv_wmma_kernel< 16,  32, 2, 3, 256, 8, true, true, true , true>
        <<<dim3(1024), dim3(256), 0, stream>>>(Q, dw2, db2, P);
    conv_wmma_kernel< 32,  64, 2, 3, 128, 8, true, true, true , true>
        <<<dim3( 256), dim3(256), 0, stream>>>(P, dw3, db3, Q);
    conv_wmma_kernel< 64, 128, 2, 3,  64, 4, true, true, false, true>
        <<<dim3( 128), dim3(128), 0, stream>>>(Q, dw4, db4, P);

    // feature convs at 64x64, 128 ch
    conv_wmma_kernel<128, 128, 1, 3, 64, 4, true, true , false, true>
        <<<dim3(128), dim3(128), 0, stream>>>(P, fw0, fb0, Q);
    conv_wmma_kernel<128, 128, 1, 3, 64, 4, true, true , false, true>
        <<<dim3(128), dim3(128), 0, stream>>>(Q, fw1, fb1, P);
    conv_wmma_kernel<128, 128, 1, 3, 64, 4, true, true , false, true>
        <<<dim3(128), dim3(128), 0, stream>>>(P, fw2, fb2, Q);
    conv_wmma_kernel<128, 128, 1, 3, 64, 4, true, false, false, true>
        <<<dim3(128), dim3(128), 0, stream>>>(Q, fw3, nullptr, P);

    // 1x1 out conv: 128 -> 16 bilateral grid, f32 NHWC out, no relu
    conv_wmma_kernel<128, 16, 1, 1, 64, 4, false, true, true, false>
        <<<dim3(128), dim3(128), 0, stream>>>(P, ow, ob, R);

    // slice + apply at 2048x2048
    slice_apply_kernel<<<dim3(2 * 2048 * 2048 / 256), dim3(256), 0, stream>>>(
        ref, R, outp);
}
